// PTPDNet_Quadratic_34411277975825
// MI455X (gfx1250) — matrix operand order repaired and checked by reading, not compiled, not run on hardware
//
#include <hip/hip_runtime.h>

typedef __attribute__((ext_vector_type(16))) _Float16 v16h;
typedef __attribute__((ext_vector_type(8)))  _Float16 v8h;
typedef __attribute__((ext_vector_type(8)))  float    v8f;
typedef int b128_t __attribute__((vector_size(16)));               // builtin param type
typedef __attribute__((address_space(3))) b128_t lds_b128_t;       // __shared__ int4

#define HSTRIDE 520   // halfs per LDS H row: 1040B = 16B aligned, 4-dword bank skew per row

// ---- f16 fragment-ordered weight workspace layout (units: halfs) ----
// Each 32x16 B-tile = 512 halfs, stored lane-major: slot = lane*16 + e,
// value = W[32*ktile + 16*(e>=8) + 8*(lane>=16) + (e&7)][16*ntile + (lane&15)]
#define W0_OFF 0                    // K padded 16->32: 1 ktile  x 32 ntiles = 32  tiles
#define W1_OFF 16384                // 16 ktiles x 32 ntiles                 = 512 tiles
#define W2_OFF (16384 + 262144)     // 16 ktiles x 10 ntiles (N 136->160)    = 160 tiles
#define WS_SLOTS (16384 + 262144 + 81920)   // 360448 halfs (~0.7 MB)

// ---- dynamic LDS layout (bytes) ----
#define LH0_OFF 0                   // 64*520*2 = 66560
#define LH1_OFF 66560               // 66560
#define LX_OFF  133120              // 64*16*4 = 4096
#define LY_OFF  137216              // 4096
#define LW_OFF  141312              // 2 x 32768 W-slab double buffer
#define LDS_BYTES (141312 + 2*32768)

// ---- gfx1250 async global->LDS copy (ASYNCcnt), with portable fallback ----
#define ASYNC_LDS (__has_builtin(__builtin_amdgcn_global_load_async_to_lds_b128) && \
                   __has_builtin(__builtin_amdgcn_s_wait_asynccnt))

__device__ __forceinline__ void async_cp16(void* lds_dst, const void* gsrc) {
#if ASYNC_LDS
    // generic LDS addr low 32 bits == raw LDS offset (aperture in high bits)
    __builtin_amdgcn_global_load_async_to_lds_b128(
        (b128_t*)gsrc,
        (lds_b128_t*)(unsigned int)(unsigned long long)lds_dst,
        0, 0);
#else
    *(v8h*)lds_dst = *(const v8h*)gsrc;          // sync bounce through VGPRs
#endif
}
__device__ __forceinline__ void async_wait0() {
#if ASYNC_LDS
    __builtin_amdgcn_s_wait_asynccnt(0);
#endif
}

__global__ __launch_bounds__(256) void prep_weights(const float* __restrict__ W0,
                                                    const float* __restrict__ W1,
                                                    const float* __restrict__ W2,
                                                    _Float16* __restrict__ ws) {
    int s = blockIdx.x * 256 + threadIdx.x;
    if (s >= WS_SLOTS) return;
    const float* W; int NT, KS, NS, rs;
    if (s < W1_OFF)            { W = W0; NT = 32; KS = 16;  NS = 512; rs = s; }
    else if (s < W2_OFF)       { W = W1; NT = 32; KS = 512; NS = 512; rs = s - W1_OFF; }
    else                       { W = W2; NT = 10; KS = 512; NS = 136; rs = s - W2_OFF; }
    int blk  = rs >> 9;
    int w    = rs & 511;
    int lane = w >> 4;
    int e    = w & 15;
    int kt   = blk / NT;
    int nt   = blk - kt * NT;
    int k    = 32 * kt + ((e & 8) << 1) + ((lane >> 4) << 3) + (e & 7);
    int n    = 16 * nt + (lane & 15);
    float v  = (k < KS && n < NS) ? W[k * NS + n] : 0.0f;
    ws[s] = (_Float16)v;
}

// Load a 16x32 f16 A-fragment from row-major LDS H (stride HSTRIDE halfs).
// ISA A layout: lane<16: e0..7 -> K 0..7, e8..15 -> K 16..23 (row = lane);
//               lane>=16: K base +8.  => two contiguous 16B chunks per lane.
__device__ __forceinline__ v16h load_a(const _Float16* lH, int rowbase, int lane, int k) {
    int row = rowbase + (lane & 15);
    int off = row * HSTRIDE + k * 32 + ((lane >> 4) << 3);
    union { v16h v; v8h h[2]; } u;
    u.h[0] = *(const v8h*)(lH + off);
    u.h[1] = *(const v8h*)(lH + off + 16);
    return u.v;
}

// D-tile (16x16 f32) -> bias + tanh -> f16 row-major LDS store.
// D layout: vgpr r, lanes 0-15: row r, col lane; lanes 16-31: row r+8, col lane-16.
__device__ __forceinline__ void store_tile_act(_Float16* lH, int rowbase, int lane,
                                               int colbase, v8f c,
                                               const float* __restrict__ bias) {
    int col  = colbase + (lane & 15);
    float bb = bias[col];
    int row0 = rowbase + ((lane >> 4) << 3);
#pragma unroll
    for (int r = 0; r < 8; ++r) {
        float x = c[r] + bb;
        float ex = __expf(2.0f * x);            // tanh(x) = 1 - 2/(e^{2x}+1)
        x = 1.0f - 2.0f / (ex + 1.0f);
        lH[(row0 + r) * HSTRIDE + col] = (_Float16)x;
    }
}

__global__ __launch_bounds__(256) void ptpd_fused(const float* __restrict__ points,
                                                  const float* __restrict__ b0,
                                                  const float* __restrict__ b1,
                                                  const float* __restrict__ b2,
                                                  const _Float16* __restrict__ ws,
                                                  float* __restrict__ out) {
    extern __shared__ char smem[];
    _Float16* lH0 = (_Float16*)(smem + LH0_OFF);
    _Float16* lH1 = (_Float16*)(smem + LH1_OFF);
    float*    lX  = (float*)(smem + LX_OFF);
    float*    lY  = (float*)(smem + LY_OFF);
    _Float16* lW0b = (_Float16*)(smem + LW_OFF);            // slab buffer 0
    _Float16* lW1b = (_Float16*)(smem + LW_OFF + 32768);    // slab buffer 1

    const _Float16* W0f = ws + W0_OFF;
    const _Float16* W1f = ws + W1_OFF;
    const _Float16* W2f = ws + W2_OFF;

    const int tid   = threadIdx.x;
    const int lane  = tid & 31;
    const int wave  = tid >> 5;
    const int msub  = wave >> 1;        // 4 row-subtiles of 16
    const int nhalf = wave & 1;         // N split in halves
    const int mrow  = msub * 16;

    // ---- init: zero Y, stage X (64 rows x 16 f32) ----
#pragma unroll
    for (int q = 0; q < 4; ++q) lY[tid * 4 + q] = 0.0f;
    {
        const float4* ps = (const float4*)(points + (size_t)blockIdx.x * 64 * 16);
        ((float4*)lX)[tid] = ps[tid];
    }
    __syncthreads();

    // ---- layer 0: X[64,16] @ W0[16,512]  (K padded to 32 with zeros) ----
    v16h aX;
    {
        int row = mrow + (lane & 15);
        int koff = (lane >> 4) << 3;
        union { v16h v; _Float16 f[16]; } u;
#pragma unroll
        for (int e = 0; e < 8; ++e) u.f[e] = (_Float16)lX[row * 16 + koff + e];
#pragma unroll
        for (int e = 8; e < 16; ++e) u.f[e] = (_Float16)0.0f;
        aX = u.v;
    }
#pragma unroll
    for (int t = 0; t < 16; ++t) {
        int ntile = nhalf * 16 + t;
        v16h b = *(const v16h*)(W0f + ntile * 512 + lane * 16);
        v8f c = {};
        c = __builtin_amdgcn_wmma_f32_16x16x32_f16(false, aX, false, b, (short)0, c, false, false);
        store_tile_act(lH0, mrow, lane, ntile * 16, c, b0);
    }

    // prefetch W1 k-slab 0 into LDS (overlaps the barrier below)
    {
        const _Float16* src = W1f;                      // k-slab 0: 16384 contiguous halfs
#pragma unroll
        for (int q = 0; q < 8; ++q) {
            int off = (q * 256 + tid) * 8;              // 16B chunk per thread per round
            async_cp16(lW0b + off, src + off);
        }
    }
    __syncthreads();                                    // lH0 ready

    // ---- layer 1: H0[64,512] @ W1[512,512], LDS-staged double-buffered slabs ----
    {
        v8f acc[16];
#pragma unroll
        for (int t = 0; t < 16; ++t) acc[t] = (v8f){};
        for (int k = 0; k < 16; ++k) {
            _Float16* cur = (k & 1) ? lW1b : lW0b;
            _Float16* nxt = (k & 1) ? lW0b : lW1b;
            async_wait0();
            __syncthreads();                            // slab k visible to all waves
            if (k < 15) {                               // DMA slab k+1 while computing k
                const _Float16* src = W1f + (k + 1) * 16384;
#pragma unroll
                for (int q = 0; q < 8; ++q) {
                    int off = (q * 256 + tid) * 8;
                    async_cp16(nxt + off, src + off);
                }
            }
            v16h a = load_a(lH0, mrow, lane, k);
#pragma unroll
            for (int t = 0; t < 16; ++t) {
                v16h b = *(const v16h*)(cur + (nhalf * 16 + t) * 512 + lane * 16);
                acc[t] = __builtin_amdgcn_wmma_f32_16x16x32_f16(false, a, false, b, (short)0,
                                                                acc[t], false, false);
            }
        }
#pragma unroll
        for (int t = 0; t < 16; ++t)
            store_tile_act(lH1, mrow, lane, nhalf * 256 + t * 16, acc[t], b1);
    }

    // prefetch W2 k-slab 0 (5120 halfs = 640 16B-chunks)
    {
        const _Float16* src = W2f;
#pragma unroll
        for (int q = 0; q < 3; ++q) {
            int c = q * 256 + tid;
            if (c < 640) async_cp16(lW0b + c * 8, src + c * 8);
        }
    }
    __syncthreads();                                    // lH1 ready

    // ---- layer 2: H1[64,512] @ W2[512,160pad] + fused tril scatter epilogue ----
    {
        v8f acc2[5];
#pragma unroll
        for (int t = 0; t < 5; ++t) acc2[t] = (v8f){};
        for (int k = 0; k < 16; ++k) {
            _Float16* cur = (k & 1) ? lW1b : lW0b;
            _Float16* nxt = (k & 1) ? lW0b : lW1b;
            async_wait0();
            __syncthreads();
            if (k < 15) {
                const _Float16* src = W2f + (k + 1) * 5120;
#pragma unroll
                for (int q = 0; q < 3; ++q) {
                    int c = q * 256 + tid;
                    if (c < 640) async_cp16(nxt + c * 8, src + c * 8);
                }
            }
            v16h a = load_a(lH1, mrow, lane, k);
#pragma unroll
            for (int t = 0; t < 5; ++t) {
                v16h b = *(const v16h*)(cur + (nhalf * 5 + t) * 512 + lane * 16);
                acc2[t] = __builtin_amdgcn_wmma_f32_16x16x32_f16(false, a, false, b, (short)0,
                                                                 acc2[t], false, false);
            }
        }
        // net[row][col] is M[i][j] (tril); y[row][j] += M[i][j] * x[row][i]
        int row0 = mrow + ((lane >> 4) << 3);
#pragma unroll
        for (int t = 0; t < 5; ++t) {
            int col = (nhalf * 5 + t) * 16 + (lane & 15);
            if (col < 136) {
                float bb = b2[col];
                int i = (int)floorf((sqrtf(8.0f * (float)col + 1.0f) - 1.0f) * 0.5f);
                while (((i + 1) * (i + 2)) / 2 <= col) ++i;
                while ((i * (i + 1)) / 2 > col) --i;
                int j = col - (i * (i + 1)) / 2;
#pragma unroll
                for (int r = 0; r < 8; ++r) {
                    int row = row0 + r;
                    float v = acc2[t][r] + bb;
                    atomicAdd(&lY[row * 16 + j], v * lX[row * 16 + i]);
                }
            }
        }
    }
    __syncthreads();

    // ---- final: vals = ||y||^2 + eps * ||x||^2 ----
    if (tid < 64) {
        int row = tid;
        float sx = 0.0f, sy = 0.0f;
#pragma unroll
        for (int j = 0; j < 16; ++j) {
            float xx = lX[row * 16 + j];
            float yy = lY[row * 16 + j];
            sx += xx * xx;
            sy += yy * yy;
        }
        out[(size_t)blockIdx.x * 64 + row] = sy + 1e-6f * sx;
    }
}

extern "C" void kernel_launch(void* const* d_in, const int* in_sizes, int n_in,
                              void* d_out, int out_size, void* d_ws, size_t ws_size,
                              hipStream_t stream) {
    const float* points = (const float*)d_in[0];
    const float* W0     = (const float*)d_in[1];
    const float* b0     = (const float*)d_in[2];
    const float* W1     = (const float*)d_in[3];
    const float* b1     = (const float*)d_in[4];
    const float* W2     = (const float*)d_in[5];
    const float* b2     = (const float*)d_in[6];
    float*       out    = (float*)d_out;
    _Float16*    wsH    = (_Float16*)d_ws;

    int rows = in_sizes[0] / 16;            // 131072

    int prep_blocks = (WS_SLOTS + 255) / 256;
    prep_weights<<<prep_blocks, 256, 0, stream>>>(W0, W1, W2, wsH);

    ptpd_fused<<<rows / 64, 256, LDS_BYTES, stream>>>(points, b0, b1, b2, wsH, out);
}
